// FactorCovModel_12060268167754
// MI455X (gfx1250) — matrix operand order repaired and checked by reading, not compiled, not run on hardware
//
#include <hip/hip_runtime.h>
#include <hip/hip_bf16.h>
#include <math.h>

// ---------------------------------------------------------------------------
// FactorCovModel on MI455X (gfx1250).
// - Dense layers (xg GEMMs, LSTM recurrence, FC): V_WMMA_F32_16X16X32_BF16
//   (bf16 in, f32 accumulate), double-buffered LDS fed by async global->LDS
//   copies (GLOBAL_LOAD_ASYNC_TO_LDS_B128 / s_wait_asynccnt) so staging of
//   chunk k+1 overlaps WMMA of chunk k.
// - Final covariance Sigma = A A^T: full f32 V_WMMA_F32_16X16X4_F32; it is
//   memory-bound (256 MB output ~= 11 us at 23.3 TB/s >> its WMMA time).
// Workspace ~162 MB carved from d_ws.
// ---------------------------------------------------------------------------

typedef float  v2f   __attribute__((ext_vector_type(2)));
typedef float  v8f   __attribute__((ext_vector_type(8)));
typedef __bf16 v8bf  __attribute__((ext_vector_type(8)));
typedef __bf16 v16bf __attribute__((ext_vector_type(16)));
typedef int    v4i   __attribute__((ext_vector_type(4)));

#define HIDDEN   512
#define GSTRIDE  2048      // 4*HIDDEN
#define BATCH    256
#define TSTEPS   64
#define INSZ     500
#define KP       512       // padded K for all bf16 GEMMs
#define NASSETS  500
#define NFACT    32
#define OUTDIM   16532     // NASSETS*NFACT + NFACT + NASSETS
#define LOFF     16000     // NASSETS*NFACT

// LDS row stride (bf16 elems): 40 elems = 80 B = 20 words; 20*r mod 64 covers
// 16 distinct banks for r=0..15 (conflict-free column reads), rows 16B-aligned.
#define LDB 40

// ---------------- async global->LDS (gfx1250) with safe fallback -----------
#if defined(__gfx1250__) && __has_builtin(__builtin_amdgcn_global_load_async_to_lds_b128)
#define ASYNC_LDS 1
#endif

#ifdef ASYNC_LDS
typedef __attribute__((address_space(1))) v4i* g_v4i_p;   // global int4*
typedef __attribute__((address_space(3))) v4i* l_v4i_p;   // LDS int4*
#endif

__device__ __forceinline__ void cp_async16(void* lds, const void* g) {
#ifdef ASYNC_LDS
  __builtin_amdgcn_global_load_async_to_lds_b128((g_v4i_p)(void*)g,
                                                 (l_v4i_p)lds, 0, 0);
#else
  *(uint4*)lds = *(const uint4*)g;
#endif
}

template <int N>
__device__ __forceinline__ void wait_async() {
#ifdef ASYNC_LDS
#if __has_builtin(__builtin_amdgcn_s_wait_asynccnt)
  __builtin_amdgcn_s_wait_asynccnt(N);
#else
  asm volatile("s_wait_asynccnt %0" ::"i"(N));
#endif
#endif
}

// ---------------- WMMA wrappers --------------------------------------------
__device__ __forceinline__ v8f wmma_bf16(v16bf a, v16bf b, v8f c) {
  return __builtin_amdgcn_wmma_f32_16x16x32_bf16(false, a, false, b, (short)0,
                                                 c, false, false);
}
__device__ __forceinline__ v8f wmma_f32_k4(v2f a, v2f b, v8f c) {
  return __builtin_amdgcn_wmma_f32_16x16x4_f32(false, a, false, b, (short)0, c,
                                               false, false);
}

// A fragment (16x32 bf16, ISA 7.12.2): lane half h: VGPR0-3 = K 8h..8h+7,
// VGPR4-7 = K 16+8h..16+8h+7 -> two contiguous 16B runs per lane.
__device__ __forceinline__ v16bf frag_a_bf16(const __bf16* base, int row0,
                                             int l16, int half) {
  const __bf16* p = base + (row0 + l16) * LDB + (half ? 8 : 0);
  v8bf lo = *(const v8bf*)p;
  v8bf hi = *(const v8bf*)(p + 16);
  return __builtin_shufflevector(lo, hi, 0, 1, 2, 3, 4, 5, 6, 7,
                                 8, 9, 10, 11, 12, 13, 14, 15);
}
// B fragment (32x16 bf16): same K order as A: lane half h: K 8h..8h+7 then
// K 16+8h..16+8h+7 (B = W^T, W row-major [N][K]) -> two 16B runs per lane.
__device__ __forceinline__ v16bf frag_b_bf16(const __bf16* base, int row0,
                                             int l16, int half) {
  const __bf16* p = base + (row0 + l16) * LDB + (half ? 8 : 0);
  v8bf lo = *(const v8bf*)p;
  v8bf hi = *(const v8bf*)(p + 16);
  return __builtin_shufflevector(lo, hi, 0, 1, 2, 3, 4, 5, 6, 7,
                                 8, 9, 10, 11, 12, 13, 14, 15);
}
// f32 16x4 fragment: (K0,K1)/(K2,K3) adjacent -> single b64 load.
__device__ __forceinline__ v2f frag_f32(const float* base, int ld, int row0,
                                        int k0, int l16, int half) {
  return *(const v2f*)(base + (size_t)(row0 + l16) * ld + k0 + (half ? 2 : 0));
}

// ---------------------------------------------------------------------------
// f32 -> bf16 with zero-padding of the K dimension to Kp.
// ---------------------------------------------------------------------------
__global__ void k_cvt_pad(const float* __restrict__ in, __bf16* __restrict__ out,
                          int R, int K, int Kp) {
  int tid = blockIdx.x * 256 + threadIdx.x;
  if (tid >= R * Kp) return;
  int r = tid / Kp, c = tid % Kp;
  out[tid] = (c < K) ? (__bf16)in[(size_t)r * K + c] : (__bf16)0.0f;
}

// ---------------------------------------------------------------------------
// C[M][N] = A[M][512]bf16 @ W[N][512]bf16^T + bias0[N] (+bias1[N]).
// 256 thr = 8 waves; block tile 64x64; each wave: 2 tiles sharing a B frag.
// Double-buffered LDS, async staging (2 async instrs / thread / stage).
// Requires M % 64 == 0; N guarded at store (OOB W rows read unguarded but
// stay inside the workspace allocation).
// ---------------------------------------------------------------------------
template <typename OutT>
__global__ void k_gemm_bf16(const __bf16* __restrict__ A,
                            const __bf16* __restrict__ W,
                            const float* __restrict__ bias0,
                            const float* __restrict__ bias1,
                            OutT* __restrict__ C, int N) {
  __shared__ __bf16 As[2][64 * LDB];
  __shared__ __bf16 Ws[2][64 * LDB];
  const int tid  = threadIdx.x;
  const int wave = tid >> 5, lane = tid & 31;
  const int half = lane >> 4, l16 = lane & 15;
  const int mt = wave >> 2, nt = wave & 3;
  const int m0 = blockIdx.y * 64;
  const int n0 = blockIdx.x * 64;
  const int sr = tid >> 2;            // staging row 0..63
  const int sc = (tid & 3) * 8;       // staging col (bf16 elems)

  const __bf16* Arow = A + (size_t)(m0 + sr) * KP + sc;
  const __bf16* Wrow = W + (size_t)(n0 + sr) * KP + sc;
  const int so = sr * LDB + sc;

  v8f acc0 = {}, acc1 = {};
  cp_async16(&As[0][so], Arow);
  cp_async16(&Ws[0][so], Wrow);
  for (int c = 0; c < 16; ++c) {
    if (c < 15) {
      cp_async16(&As[(c + 1) & 1][so], Arow + (c + 1) * 32);
      cp_async16(&Ws[(c + 1) & 1][so], Wrow + (c + 1) * 32);
      wait_async<2>();                 // chunk c landed; c+1 may be in flight
    } else {
      wait_async<0>();
    }
    __syncthreads();
    const __bf16* ab = As[c & 1];
    const __bf16* wb = Ws[c & 1];
    v16bf a0 = frag_a_bf16(ab, mt * 16, l16, half);
    v16bf a1 = frag_a_bf16(ab, mt * 16 + 32, l16, half);
    v16bf b  = frag_b_bf16(wb, nt * 16, l16, half);
    acc0 = wmma_bf16(a0, b, acc0);
    acc1 = wmma_bf16(a1, b, acc1);
    __syncthreads();                   // safe to overwrite this buffer later
  }
  const int col = n0 + nt * 16 + l16;
  if (col < N) {
    float bv = bias0[col] + (bias1 ? bias1[col] : 0.0f);
    #pragma unroll
    for (int j = 0; j < 8; ++j) {
      int r0 = m0 + mt * 16 + j + (half ? 8 : 0);
      C[(size_t)r0 * N + col]        = (OutT)(acc0[j] + bv);
      C[(size_t)(r0 + 32) * N + col] = (OutT)(acc1[j] + bv);
    }
  }
}

// ---------------------------------------------------------------------------
// One fused LSTM step for a 32(batch) x 16(hidden) tile, all 4 gates.
// 128 thr = 4 waves; wave g computes gate g (2 WMMAs per K-chunk, shared B).
// Async double-buffered staging (3 async instrs / thread / stage).
// Rows of xg are batch-major: (b*TSTEPS + t).
// ---------------------------------------------------------------------------
__global__ void k_lstm_step(const __bf16* __restrict__ xg,    // [B*T][2048]
                            const __bf16* __restrict__ w_hh,  // [2048][512]
                            const __bf16* __restrict__ h_prev, int hp_stride,
                            float* __restrict__ c_io,         // [256][512] f32
                            __bf16* __restrict__ h_out, int ho_stride,
                            int t) {
  __shared__ __bf16 As[2][32 * LDB];
  __shared__ __bf16 Ws[2][64 * LDB];
  __shared__ float  Gs[4 * 32 * 16];
  const int tid  = threadIdx.x;        // 128
  const int g    = tid >> 5, lane = tid & 31;
  const int half = lane >> 4, l16 = lane & 15;
  const int m0   = blockIdx.x * 32;    // batch rows
  const int n0   = blockIdx.y * 16;    // hidden cols

  // staging addressing: As 32 rows x 4 groups (1 instr/thread);
  // Ws 64 rows (gate*16+r) x 4 groups (2 instrs/thread).
  const int ar = tid >> 2, ac = (tid & 3) * 8;
  const __bf16* hrow = h_prev + (size_t)(m0 + ar) * hp_stride + ac;
  const int ao = ar * LDB + ac;
  const int wg0 = tid >> 6, wr0 = (tid >> 2) & 15, wc0 = (tid & 3) * 8;
  const __bf16* wrow0 = w_hh + (size_t)(wg0 * HIDDEN + n0 + wr0) * KP + wc0;
  const __bf16* wrow1 = wrow0 + (size_t)2 * HIDDEN * KP;   // gates 2..3
  const int wo0 = (wg0 * 16 + wr0) * LDB + wc0;
  const int wo1 = wo0 + 32 * LDB;

  v8f acc0 = {}, acc1 = {};
  cp_async16(&As[0][ao], hrow);
  cp_async16(&Ws[0][wo0], wrow0);
  cp_async16(&Ws[0][wo1], wrow1);
  for (int c = 0; c < 16; ++c) {
    if (c < 15) {
      int nb = (c + 1) & 1, k0 = (c + 1) * 32;
      cp_async16(&As[nb][ao], hrow + k0);
      cp_async16(&Ws[nb][wo0], wrow0 + k0);
      cp_async16(&Ws[nb][wo1], wrow1 + k0);
      wait_async<3>();
    } else {
      wait_async<0>();
    }
    __syncthreads();
    const __bf16* ab = As[c & 1];
    const __bf16* wb = Ws[c & 1] + g * 16 * LDB;
    v16bf a0 = frag_a_bf16(ab, 0, l16, half);
    v16bf a1 = frag_a_bf16(ab, 16, l16, half);
    v16bf b  = frag_b_bf16(wb, 0, l16, half);
    acc0 = wmma_bf16(a0, b, acc0);
    acc1 = wmma_bf16(a1, b, acc1);
    __syncthreads();
  }
  // add precomputed xg (batch-major rows) and stash gate tiles in LDS
  #pragma unroll
  for (int j = 0; j < 8; ++j) {
    int r = j + (half ? 8 : 0);
    int row = m0 + r;
    Gs[(g * 32 + r) * 16 + l16] = acc0[j] +
        (float)xg[((size_t)row * TSTEPS + t) * GSTRIDE + g * HIDDEN + n0 + l16];
    Gs[(g * 32 + r + 16) * 16 + l16] = acc1[j] +
        (float)xg[((size_t)(row + 16) * TSTEPS + t) * GSTRIDE + g * HIDDEN + n0 + l16];
  }
  __syncthreads();
  #pragma unroll
  for (int ii = 0; ii < 4; ++ii) {       // pointwise: 512 elems / 128 thr
    int e = tid + ii * 128;
    int r = e >> 4, cc = e & 15;
    float gi = Gs[(0 * 32 + r) * 16 + cc];
    float gf = Gs[(1 * 32 + r) * 16 + cc];
    float gg = Gs[(2 * 32 + r) * 16 + cc];
    float go = Gs[(3 * 32 + r) * 16 + cc];
    float si = 1.0f / (1.0f + __expf(-gi));
    float sf = 1.0f / (1.0f + __expf(-gf));
    float so = 1.0f / (1.0f + __expf(-go));
    float tg = tanhf(gg);
    int row = m0 + r, col = n0 + cc;
    size_t cidx = (size_t)row * HIDDEN + col;
    float cn = sf * c_io[cidx] + si * tg;
    c_io[cidx] = cn;
    h_out[(size_t)row * ho_stride + col] = (__bf16)(so * tanhf(cn));
  }
}

// ---------------------------------------------------------------------------
// Ascaled[b][n][f] = (n<500 ? raw[b][n*32+f] : 0) * sqrt(exp(raw[b][16000+f]))
// idio[b][n]      = exp(raw[b][16032+n])
// ---------------------------------------------------------------------------
__global__ void k_prep(const float* __restrict__ raw,
                       float* __restrict__ Ascaled,
                       float* __restrict__ idio) {
  int tid = blockIdx.x * 256 + threadIdx.x;   // BATCH*512*32 threads
  int b = tid >> 14;
  int r = (tid >> 5) & 511;
  int f = tid & 31;
  const float* rb = raw + (size_t)b * OUTDIM;
  float sv = sqrtf(__expf(rb[LOFF + f]));
  float lv = (r < NASSETS) ? rb[r * NFACT + f] : 0.0f;
  Ascaled[tid] = lv * sv;
  if (tid < BATCH * NASSETS) {
    int bb = tid / NASSETS, n = tid % NASSETS;
    idio[tid] = __expf(raw[(size_t)bb * OUTDIM + LOFF + NFACT + n]);
  }
}

// ---------------------------------------------------------------------------
// Sigma[b] = A A^T (+ diag(idio)); A = Ascaled[b] (512x32 f32, rows>=500 = 0).
// Full f32 WMMA; 256 thr = 8 waves; 64x64 block; K=32 staged once (async).
// ---------------------------------------------------------------------------
__global__ void k_sigma(const float* __restrict__ Asc,   // [B][512][32]
                        const float* __restrict__ idio,  // [B][500]
                        float* __restrict__ out) {       // [B][500][500]
  __shared__ float Ar[64 * 36];
  __shared__ float Ac[64 * 36];
  const int tid  = threadIdx.x;
  const int wave = tid >> 5, lane = tid & 31;
  const int half = lane >> 4, l16 = lane & 15;
  const int cb = blockIdx.x, rb = blockIdx.y, b = blockIdx.z;
  const float* Ab = Asc + (size_t)b * 512 * 32;

  #pragma unroll
  for (int ii = 0; ii < 2; ++ii) {     // 512 b128 per panel, 4 instrs/thread
    int i = tid + ii * 256;
    int r = i >> 3, c4 = (i & 7) * 4;
    cp_async16(Ar + r * 36 + c4, Ab + (size_t)(rb * 64 + r) * 32 + c4);
    cp_async16(Ac + r * 36 + c4, Ab + (size_t)(cb * 64 + r) * 32 + c4);
  }
  wait_async<0>();
  __syncthreads();

  for (int tile = wave; tile < 16; tile += 8) {   // 2 tiles per wave
    int rt = tile >> 2, ct = tile & 3;
    v8f acc = {};
    #pragma unroll
    for (int kk = 0; kk < 32; kk += 4) {
      v2f a  = frag_f32(Ar, 36, rt * 16, kk, l16, half);
      v2f bf = frag_f32(Ac, 36, ct * 16, kk, l16, half);
      acc = wmma_f32_k4(a, bf, acc);
    }
    int col = cb * 64 + ct * 16 + l16;
    #pragma unroll
    for (int j = 0; j < 8; ++j) {
      int row = rb * 64 + rt * 16 + j + (half ? 8 : 0);
      if (row < NASSETS && col < NASSETS) {
        float v = acc[j];
        if (row == col) v += idio[(size_t)b * NASSETS + row];
        out[((size_t)b * NASSETS + row) * NASSETS + col] = v;
      }
    }
  }
}

// ---------------------------------------------------------------------------
extern "C" void kernel_launch(void* const* d_in, const int* in_sizes, int n_in,
                              void* d_out, int out_size, void* d_ws, size_t ws_size,
                              hipStream_t stream) {
  const float* x     = (const float*)d_in[0];
  const float* w_ih0 = (const float*)d_in[1];
  const float* w_hh0 = (const float*)d_in[2];
  const float* b_ih0 = (const float*)d_in[3];
  const float* b_hh0 = (const float*)d_in[4];
  const float* w_ih1 = (const float*)d_in[5];
  const float* w_hh1 = (const float*)d_in[6];
  const float* b_ih1 = (const float*)d_in[7];
  const float* b_hh1 = (const float*)d_in[8];
  const float* fc_w  = (const float*)d_in[9];
  const float* fc_b  = (const float*)d_in[10];
  float* out = (float*)d_out;

  char* p = (char*)d_ws;
  auto carveB = [&](size_t bytes) {
    char* q = p;
    p += (bytes + 255) & ~(size_t)255;
    return q;
  };
  const int M = BATCH * TSTEPS;                                // 16384
  __bf16* x_bf    = (__bf16*)carveB((size_t)M * KP * 2);
  __bf16* wih0_bf = (__bf16*)carveB((size_t)GSTRIDE * KP * 2);
  __bf16* whh0_bf = (__bf16*)carveB((size_t)GSTRIDE * KP * 2);
  __bf16* wih1_bf = (__bf16*)carveB((size_t)GSTRIDE * KP * 2);
  __bf16* whh1_bf = (__bf16*)carveB((size_t)GSTRIDE * KP * 2);
  __bf16* fcw_bf  = (__bf16*)carveB((size_t)OUTDIM * KP * 2);
  __bf16* xg_bf   = (__bf16*)carveB((size_t)M * GSTRIDE * 2);  // 67 MB
  __bf16* hall_bf = (__bf16*)carveB((size_t)M * HIDDEN * 2);
  __bf16* hz      = (__bf16*)carveB((size_t)BATCH * HIDDEN * 2);
  __bf16* hA_bf   = (__bf16*)carveB((size_t)BATCH * HIDDEN * 2);
  __bf16* hB_bf   = (__bf16*)carveB((size_t)BATCH * HIDDEN * 2);
  float*  cbuf    = (float*)carveB((size_t)BATCH * HIDDEN * 4);
  float*  raw     = (float*)carveB((size_t)BATCH * OUTDIM * 4);
  float*  Asc     = (float*)carveB((size_t)BATCH * 512 * NFACT * 4);
  float*  idio    = (float*)carveB((size_t)BATCH * NASSETS * 4);

  auto cvt = [&](const float* src, __bf16* dst, int R, int K) {
    int n = R * KP;
    k_cvt_pad<<<(n + 255) / 256, 256, 0, stream>>>(src, dst, R, K, KP);
  };
  cvt(x,     x_bf,    M,       INSZ);
  cvt(w_ih0, wih0_bf, GSTRIDE, INSZ);
  cvt(w_hh0, whh0_bf, GSTRIDE, HIDDEN);
  cvt(w_ih1, wih1_bf, GSTRIDE, HIDDEN);
  cvt(w_hh1, whh1_bf, GSTRIDE, HIDDEN);
  cvt(fc_w,  fcw_bf,  OUTDIM,  HIDDEN);

  (void)hipMemsetAsync(hz, 0, (size_t)BATCH * HIDDEN * 2, stream);

  // ---- Layer 0: xg = x @ w_ih0^T + b_ih0 + b_hh0 (bf16 out) ----
  k_gemm_bf16<__bf16><<<dim3(GSTRIDE / 64, M / 64), 256, 0, stream>>>(
      x_bf, wih0_bf, b_ih0, b_hh0, xg_bf, GSTRIDE);
  (void)hipMemsetAsync(cbuf, 0, (size_t)BATCH * HIDDEN * 4, stream);
  for (int t = 0; t < TSTEPS; ++t) {
    const __bf16* hp = (t == 0) ? hz : (hall_bf + (size_t)(t - 1) * HIDDEN);
    int hps = (t == 0) ? HIDDEN : TSTEPS * HIDDEN;
    k_lstm_step<<<dim3(BATCH / 32, HIDDEN / 16), 128, 0, stream>>>(
        xg_bf, whh0_bf, hp, hps, cbuf,
        hall_bf + (size_t)t * HIDDEN, TSTEPS * HIDDEN, t);
  }

  // ---- Layer 1: xg = h_all @ w_ih1^T + b_ih1 + b_hh1 (xg reused) ----
  k_gemm_bf16<__bf16><<<dim3(GSTRIDE / 64, M / 64), 256, 0, stream>>>(
      hall_bf, wih1_bf, b_ih1, b_hh1, xg_bf, GSTRIDE);
  (void)hipMemsetAsync(cbuf, 0, (size_t)BATCH * HIDDEN * 4, stream);
  __bf16* hp = hA_bf;
  __bf16* hn = hB_bf;
  for (int t = 0; t < TSTEPS; ++t) {
    const __bf16* hsrc = (t == 0) ? hz : hp;
    k_lstm_step<<<dim3(BATCH / 32, HIDDEN / 16), 128, 0, stream>>>(
        xg_bf, whh1_bf, hsrc, HIDDEN, cbuf, hn, HIDDEN, t);
    __bf16* tmp = hp; hp = hn; hn = tmp;
  }
  // hp now holds h at t = 63

  // ---- FC: raw = h_last @ fc_w^T + fc_b (f32 out) ----
  k_gemm_bf16<float><<<dim3((OUTDIM + 63) / 64, BATCH / 64), 256, 0, stream>>>(
      hp, fcw_bf, fc_b, nullptr, raw, OUTDIM);

  // ---- Scale loadings by sqrt(factor_vars); materialize idio vars ----
  k_prep<<<(BATCH * 512 * NFACT) / 256, 256, 0, stream>>>(raw, Asc, idio);

  // ---- Sigma = A A^T + diag(idio): dominant 4.1 GFLOP, full f32 ----
  k_sigma<<<dim3(8, 8, BATCH), 256, 0, stream>>>(Asc, idio, out);
}
